// SutraV05_34084860461129
// MI455X (gfx1250) — matrix operand order repaired and checked by reading, not compiled, not run on hardware
//
#include <hip/hip_runtime.h>
#include <math.h>

// ---------------- constants ----------------
constexpr int cB  = 2;
constexpr int cT  = 1024;
constexpr int cBT = 2048;
constexpr int cD  = 768;
constexpr int cF  = 1536;
constexpr int cH  = 256;
constexpr int cV  = 50257;
constexpr int cVpad = 50304;           // next multiple of 64
constexpr int cS  = 7;
constexpr int cKRET = 8;
constexpr int cSTEPS = 8;

typedef __attribute__((ext_vector_type(16))) _Float16 v16h;
typedef __attribute__((ext_vector_type(8)))  _Float16 v8h;
typedef __attribute__((ext_vector_type(2)))  __fp16   v2hf;   // cvt_pkrtz result type
typedef __attribute__((ext_vector_type(8)))  float    v8f;

__device__ const unsigned char c_graph[49] = {
  1,1,1,0,0,0,0,
  0,1,1,1,0,0,0,
  0,0,1,1,1,0,0,
  0,0,0,1,1,1,1,
  0,0,0,1,1,1,1,
  0,0,0,1,0,1,1,
  0,0,0,1,0,0,1};

__device__ __forceinline__ float dev_silu(float x) {
  return x * (1.0f / (1.0f + __expf(-x)));
}
__device__ __forceinline__ float dev_softplus(float x) {
  return (x > 20.0f) ? x : log1pf(__expf(x));
}

// ---------------- generic WMMA GEMM (double-buffered, async B) ----------------
// C[M,N] = act( scale * (A[M,K] @ B[K,N]) + bias + (accumulate ? C : 0) )
// Contract (all call sites satisfy it):
//   M % 64 == 0, K % 32 == 0, lda % 4 == 0, ldb % 8 == 0
//   B has at least ceil(N/64)*64 valid rows/cols (emb is padded to 50304)
// A f32 row-major. B f16: K x N row-major, or (btrans) N x K row-major.
__global__ __launch_bounds__(128)
void gemm_f16w(const float* __restrict__ A, const _Float16* __restrict__ Bh,
               float* __restrict__ C, int M, int N, int K,
               int lda, int ldb, int ldc,
               long long sAb, long long sBb, long long sCb,
               const float* __restrict__ bias, float scale,
               int accumulate, int act, int btrans)
{
  __shared__ __align__(16) _Float16 shA[2][64][32];   // [buf][m][k]
  __shared__ __align__(16) _Float16 shB[2][64][32];   // [buf][n][k]

  const int tid  = threadIdx.x;
  const int wave = tid >> 5;
  const int lane = tid & 31;
  const int hl   = lane & 15;
  const int hi   = lane >> 4;

  const int z = blockIdx.z;
  const float*    Ab = A  + (long long)z * sAb;
  const _Float16* Bb = Bh + (long long)z * sBb;
  float*          Cb = C  + (long long)z * sCb;

  const int m0 = blockIdx.y * 64;
  const int n0 = blockIdx.x * 64;

  // ---- K-invariant staging geometry ----
  int ar[4], ac[4];
  const float* aptr[4];
  #pragma unroll
  for (int i = 0; i < 4; ++i) {
    int c4 = tid + i * 128;            // 512 float4 chunks: 64 rows x 8
    ar[i] = c4 >> 3;
    ac[i] = (c4 & 7) * 4;
    aptr[i] = Ab + (long long)(m0 + ar[i]) * lda + ac[i];
  }
  int bn[2], bk[2];
  const _Float16* bptr[2];
  #pragma unroll
  for (int i = 0; i < 2; ++i) {
    int c8 = tid + i * 128;
    if (btrans) { bn[i] = c8 >> 2; bk[i] = (c8 & 3) * 8; }
    else        { bk[i] = c8 >> 3; bn[i] = (c8 & 7) * 8; }
    bptr[i] = btrans ? (Bb + (long long)(n0 + bn[i]) * ldb + bk[i])
                     : (Bb + (long long)bk[i] * ldb + (n0 + bn[i]));
  }

  auto preloadA = [&](int k0, float4* f) {
    #pragma unroll
    for (int i = 0; i < 4; ++i)
      f[i] = *(const float4*)(aptr[i] + k0);
  };
  auto commitA = [&](int buf, const float4* f) {
    #pragma unroll
    for (int i = 0; i < 4; ++i) {
      v2hf lo  = __builtin_amdgcn_cvt_pkrtz(f[i].x, f[i].y);
      v2hf hi2 = __builtin_amdgcn_cvt_pkrtz(f[i].z, f[i].w);
      *(v2hf*)&shA[buf][ar[i]][ac[i]]     = lo;
      *(v2hf*)&shA[buf][ar[i]][ac[i] + 2] = hi2;
    }
  };
  auto preloadB = [&](int k0, v8h* q) {
    #pragma unroll
    for (int i = 0; i < 2; ++i)
      q[i] = *(const v8h*)(bptr[i] + (long long)k0 * ldb);
  };
  auto commitB = [&](int buf, const v8h* q) {
    #pragma unroll
    for (int i = 0; i < 2; ++i)
      #pragma unroll
      for (int e = 0; e < 8; ++e) shB[buf][bn[i] + e][bk[i]] = q[i][e];
  };
  // pure 16B copies, no conversion: CDNA5 async global->LDS DMA (ASYNCcnt)
  auto asyncB = [&](int k0, int buf) {
    #pragma unroll
    for (int i = 0; i < 2; ++i) {
      unsigned ldsoff = (unsigned)(size_t)&shB[buf][bn[i]][bk[i]];
      unsigned long long gaddr = (unsigned long long)(bptr[i] + k0);
      asm volatile("global_load_async_to_lds_b128 %0, %1, off"
                   :: "v"(ldsoff), "v"(gaddr) : "memory");
    }
  };

  v8f acc[4];
  #pragma unroll
  for (int i = 0; i < 4; ++i)
    #pragma unroll
    for (int j = 0; j < 8; ++j) acc[i][j] = 0.0f;

  const int nk = K >> 5;
  float4 fA[4];
  v8h qB[2];

  // ---- prologue: stage tile 0 into buffer 0 ----
  preloadA(0, fA);
  if (btrans) asyncB(0, 0); else preloadB(0, qB);
  commitA(0, fA);
  if (btrans) asm volatile("s_wait_asynccnt 0" ::: "memory");
  else commitB(0, qB);
  __syncthreads();

  // ---- pipelined main loop ----
  for (int kt = 0; kt < nk; ++kt) {
    const int cur = kt & 1, nxt = cur ^ 1;
    const bool more = (kt + 1 < nk);
    if (more) {
      preloadA((kt + 1) << 5, fA);                 // loads in flight over WMMA
      if (btrans) asyncB((kt + 1) << 5, nxt);      // DMA in flight over WMMA
      else preloadB((kt + 1) << 5, qB);
      if (kt + 2 < nk)
        __builtin_prefetch(aptr[0] + ((kt + 2) << 5), 0, 1);
    }

    // ---- fragments in contiguous k order (k = 16*hi + e), same for A and B ----
    v16h af;
    #pragma unroll
    for (int e = 0; e < 16; ++e) {
      int kk = e + hi * 16;
      af[e] = shA[cur][wave * 16 + hl][kk];
    }
    #pragma unroll
    for (int nb = 0; nb < 4; ++nb) {
      v16h bf;
      #pragma unroll
      for (int e = 0; e < 16; ++e)
        bf[e] = shB[cur][nb * 16 + hl][e + hi * 16];
      acc[nb] = __builtin_amdgcn_wmma_f32_16x16x32_f16(
          false, af, false, bf, (short)0, acc[nb], false, false);
    }

    if (more) {
      commitA(nxt, fA);
      if (btrans) asm volatile("s_wait_asynccnt 0" ::: "memory");
      else commitB(nxt, qB);
    }
    __syncthreads();
  }

  // ---- epilogue (store guard only on N tail; M always tile-aligned) ----
  #pragma unroll
  for (int nb = 0; nb < 4; ++nb) {
    int n = n0 + nb * 16 + hl;
    if (n >= N) continue;
    #pragma unroll
    for (int r = 0; r < 8; ++r) {
      int m = m0 + wave * 16 + hi * 8 + r;
      float v = acc[nb][r] * scale;
      if (bias) v += bias[n];
      long long off = (long long)m * ldc + n;
      if (accumulate) v += Cb[off];
      if (act == 1)      v = dev_silu(v);
      else if (act == 2) v = dev_softplus(v) + 0.1f;
      Cb[off] = v;
    }
  }
}

// ---------------- small kernels ----------------
// vectorized f32 -> f16 (8 per thread); n must be a multiple of 8.
__global__ void k_convert8(_Float16* dst, const float* src, long long n8) {
  long long i = (long long)blockIdx.x * blockDim.x + threadIdx.x;
  if (i >= n8) return;
  const float4* s = (const float4*)(src + i * 8);
  float4 a = s[0], b = s[1];
  v2hf p0 = __builtin_amdgcn_cvt_pkrtz(a.x, a.y);
  v2hf p1 = __builtin_amdgcn_cvt_pkrtz(a.z, a.w);
  v2hf p2 = __builtin_amdgcn_cvt_pkrtz(b.x, b.y);
  v2hf p3 = __builtin_amdgcn_cvt_pkrtz(b.z, b.w);
  v2hf* d = (v2hf*)(dst + i * 8);
  d[0] = p0; d[1] = p1; d[2] = p2; d[3] = p3;
}

__global__ void k_zero_h(_Float16* dst, long long n) {
  long long i = (long long)blockIdx.x * blockDim.x + threadIdx.x;
  if (i < n) dst[i] = (_Float16)0.0f;
}

__global__ void k_embed(const int* x, const float* emb, const float* pos, float* h) {
  long long i = (long long)blockIdx.x * blockDim.x + threadIdx.x;
  if (i >= (long long)cBT * cD) return;
  int g = (int)(i / cD), d = (int)(i % cD);
  h[i] = emb[(long long)x[g] * cD + d] + pos[(long long)(g % cT) * cD + d];
}

__global__ void k_pi_init(float* pi) {
  int i = blockIdx.x * blockDim.x + threadIdx.x;
  if (i < cBT * cS) pi[i] = ((i % cS) == 2) ? 1.0f : 0.0f;
}

// raw49 = hidden(BT,256) @ W2(256,49) + b2
__global__ void k_mm49(const float* hid, const float* W2, const float* b2, float* out) {
  int i = blockIdx.x * blockDim.x + threadIdx.x;
  if (i >= cBT * 49) return;
  int t = i / 49, j = i % 49;
  float s = b2[j];
  const float* hr = hid + (long long)t * cH;
  for (int k = 0; k < cH; ++k) s += hr[k] * W2[k * 49 + j];
  out[i] = s;
}

__global__ void k_transition(const float* raw, const float* pi, float* pi_ev) {
  int t = blockIdx.x * blockDim.x + threadIdx.x;
  if (t >= cBT) return;
  float pe[7] = {0, 0, 0, 0, 0, 0, 0};
  for (int s = 0; s < 7; ++s) {
    float row[7], mx = -1e30f;
    for (int r = 0; r < 7; ++r) {
      float v = c_graph[s * 7 + r] ? raw[t * 49 + s * 7 + r] : -1e30f;
      row[r] = v; mx = fmaxf(mx, v);
    }
    float sum = 0.0f;
    for (int r = 0; r < 7; ++r) { row[r] = __expf(row[r] - mx); sum += row[r]; }
    float ps = pi[t * 7 + s], inv = 1.0f / sum;
    for (int r = 0; r < 7; ++r) pe[r] += ps * row[r] * inv;
  }
  for (int r = 0; r < 7; ++r) pi_ev[t * 7 + r] = pe[r];
}

__global__ void k_stage_acc(float* weighted, const float* sout, const float* pi,
                            int s, int first) {
  long long i = (long long)blockIdx.x * blockDim.x + threadIdx.x;
  if (i >= (long long)cBT * cD) return;
  int t = (int)(i / cD);
  float w = pi[t * 7 + s] * sout[i];
  weighted[i] = first ? w : (weighted[i] + w);
}

__global__ void k_evidence(const float* weighted, const float* evW, const float* evb,
                           float* evid) {
  int i = blockIdx.x * blockDim.x + threadIdx.x;
  if (i >= cBT * cS) return;
  int t = i / cS, s = i % cS;
  float acc = evb[s];
  const float* wr = weighted + (long long)t * cD;
  for (int d = 0; d < cD; ++d) acc += wr[d] * evW[d * cS + s];
  evid[i] = acc;
}

__global__ void k_pi_update(const float* pi_ev, const float* evid, float* pi) {
  int t = blockIdx.x * blockDim.x + threadIdx.x;
  if (t >= cBT) return;
  float pe[7], ev[7];
  float mx = -1e30f;
  for (int r = 0; r < 7; ++r) {
    pe[r] = pi_ev[t * 7 + r];
    ev[r] = 2.0f * evid[t * 7 + r];
    mx = fmaxf(mx, ev[r]);
  }
  float sum = 0.0f, sm[7];
  for (int r = 0; r < 7; ++r) { sm[r] = __expf(ev[r] - mx); sum += sm[r]; }
  float inv = 1.0f / sum, s2 = 0.0f, pn[7];
  for (int r = 0; r < 7; ++r) { pn[r] = pe[r] * sm[r] * inv; s2 += pn[r]; }
  float inv2 = 1.0f / fmaxf(s2, 1e-8f);
  for (int r = 0; r < 7; ++r) pn[r] *= inv2;
  int i1 = 0;
  for (int r = 1; r < 7; ++r) if (pn[r] > pn[i1]) i1 = r;
  int i2 = (i1 == 0) ? 1 : 0;
  for (int r = 0; r < 7; ++r) if (r != i1 && pn[r] > pn[i2]) i2 = r;
  float tot = pn[i1] + pn[i2], inv3 = 1.0f / fmaxf(tot, 1e-8f);
  for (int r = 0; r < 7; ++r) pi[t * 7 + r] = 0.0f;
  pi[t * 7 + i1] = pn[i1] * inv3;
  pi[t * 7 + i2] = pn[i2] * inv3;
}

__global__ void k_shift_silu(const float* X1, const float* X2, float* out, int w) {
  long long i = (long long)blockIdx.x * blockDim.x + threadIdx.x;
  if (i >= (long long)cBT * cD) return;
  int t = (int)(i / cD);
  int tt = t % cT;
  float s = X1[i];
  if (tt >= w) s += X2[i - (long long)w * cD];
  out[i] = dev_silu(s);
}

__global__ void k_local_fix(float* local, const float* b2) {
  long long i = (long long)blockIdx.x * blockDim.x + threadIdx.x;
  if (i >= (long long)cBT * cD) return;
  int d = (int)(i % cD);
  local[i] = local[i] * 0.25f + b2[d];
}

__global__ void k_topk(const float* scores, float* attn, int* tidx) {
  int g = blockIdx.x * blockDim.x + threadIdx.x;
  if (g >= cBT) return;
  int b = g / cT, i = g % cT;
  const float* row = scores + (long long)b * cT * cT + (long long)i * cT;
  float tv[cKRET]; int ti[cKRET];
  for (int k = 0; k < cKRET; ++k) { tv[k] = -1e30f; ti[k] = 0; }
  for (int j = 0; j <= i; ++j) {
    float v = row[j];
    if (v > tv[cKRET - 1]) {
      int p = cKRET - 1;
      while (p > 0 && v > tv[p - 1]) { tv[p] = tv[p - 1]; ti[p] = ti[p - 1]; --p; }
      tv[p] = v; ti[p] = j;
    }
  }
  float mx = tv[0], sum = 0.0f, e[cKRET];
  for (int k = 0; k < cKRET; ++k) { e[k] = __expf(tv[k] - mx); sum += e[k]; }
  float inv = 1.0f / sum;
  for (int k = 0; k < cKRET; ++k) { attn[g * cKRET + k] = e[k] * inv; tidx[g * cKRET + k] = ti[k]; }
}

__global__ void k_gmsg(const float* attn, const int* tidx, const float* v, float* gmsg) {
  long long i = (long long)blockIdx.x * blockDim.x + threadIdx.x;
  if (i >= (long long)cBT * cD) return;
  int t = (int)(i / cD), d = (int)(i % cD);
  int b = t / cT;
  float s = 0.0f;
  for (int k = 0; k < cKRET; ++k) {
    int j = tidx[t * cKRET + k];
    s += attn[t * cKRET + k] * v[((long long)(b * cT + j)) * cD + d];
  }
  gmsg[i] = s;
}

__global__ void k_msg_scale(float* messages, const float* pi) {
  long long i = (long long)blockIdx.x * blockDim.x + threadIdx.x;
  if (i >= (long long)cBT * cD) return;
  int t = (int)(i / cD);
  messages[i] *= pi[t * 7 + 3];
}

__global__ void k_bayes(float* mu, float* lam, const float* mraw, const float* kraw,
                        const float* pi, const float* weighted) {
  long long i = (long long)blockIdx.x * blockDim.x + threadIdx.x;
  if (i >= (long long)cBT * cD) return;
  int t = (int)(i / cD);
  float kap = dev_softplus(kraw[i]);
  float eff = fminf(pi[t * 7 + 4] * kap, 10.0f);
  float l = lam[i];
  float ln = l + eff;
  float mu_new = (l * mu[i] + eff * mraw[i]) / fmaxf(ln, 1e-6f);
  lam[i] = ln;
  mu[i] = mu_new + 0.1f * weighted[i];
}

__global__ void k_layernorm(const float* mu, const float* g, const float* b, float* fin) {
  int t = blockIdx.x * blockDim.x + threadIdx.x;
  if (t >= cBT) return;
  const float* row = mu + (long long)t * cD;
  float mean = 0.0f;
  for (int d = 0; d < cD; ++d) mean += row[d];
  mean /= cD;
  float var = 0.0f;
  for (int d = 0; d < cD; ++d) { float df = row[d] - mean; var += df * df; }
  var /= cD;
  float inv = rsqrtf(var + 1e-5f);
  float* out = fin + (long long)t * cD;
  for (int d = 0; d < cD; ++d) out[d] = (row[d] - mean) * inv * g[d] + b[d];
}

// ---------------- host orchestration ----------------
static inline char* carve(char*& p, size_t bytes) {
  char* r = p;
  p += (bytes + 255) & ~(size_t)255;
  return r;
}

static void gemm(hipStream_t st, const float* A, const _Float16* Bh, float* C,
                 int M, int N, int K, int lda, int ldb, int ldc,
                 const float* bias, float scale, int acc, int act, int btrans,
                 int Z = 1, long long sA = 0, long long sB = 0, long long sC = 0) {
  dim3 g((N + 63) / 64, (M + 63) / 64, Z), b(128);
  gemm_f16w<<<g, b, 0, st>>>(A, Bh, C, M, N, K, lda, ldb, ldc, sA, sB, sC,
                             bias, scale, acc, act, btrans);
}

#define EWG(n) dim3((unsigned)(((n) + 255) / 256)), dim3(256)

extern "C" void kernel_launch(void* const* d_in, const int* in_sizes, int n_in,
                              void* d_out, int out_size, void* d_ws, size_t ws_size,
                              hipStream_t stream) {
  (void)in_sizes; (void)n_in; (void)out_size; (void)ws_size;

  // ---- inputs (setup_inputs dict order) ----
  const int*   x        = (const int*)  d_in[0];
  const float* emb      = (const float*)d_in[1];
  const float* pos_emb  = (const float*)d_in[2];
  const float* init_muW = (const float*)d_in[3];
  const float* init_lmW = (const float*)d_in[4];
  const float* transW1  = (const float*)d_in[5];
  const float* transW2  = (const float*)d_in[6];
  const float* stageW1  = (const float*)d_in[7];
  const float* stageW2  = (const float*)d_in[8];
  const float* evW      = (const float*)d_in[9];
  const float* qW       = (const float*)d_in[10];
  const float* kW       = (const float*)d_in[11];
  const float* vW       = (const float*)d_in[12];
  const float* msgW1    = (const float*)d_in[13];
  const float* msgW2    = (const float*)d_in[14];
  const float* outW     = (const float*)d_in[15];
  const float* wmsgW    = (const float*)d_in[16];
  const float* wgainW   = (const float*)d_in[17];
  const float* init_mub = (const float*)d_in[18];
  const float* init_lmb = (const float*)d_in[19];
  const float* transb1  = (const float*)d_in[20];
  const float* transb2  = (const float*)d_in[21];
  const float* stageb1  = (const float*)d_in[22];
  const float* stageb2  = (const float*)d_in[23];
  const float* evb      = (const float*)d_in[24];
  const float* qb       = (const float*)d_in[25];
  const float* kb       = (const float*)d_in[26];
  const float* vb       = (const float*)d_in[27];
  const float* msgb1    = (const float*)d_in[28];
  const float* msgb2    = (const float*)d_in[29];
  const float* outb     = (const float*)d_in[30];
  const float* wmsgb    = (const float*)d_in[31];
  const float* wgainb   = (const float*)d_in[32];
  const float* lng      = (const float*)d_in[33];
  const float* lnb      = (const float*)d_in[34];

  float* logits = (float*)d_out;

  // ---- workspace carve ----
  char* p = (char*)d_ws;
  _Float16* emb_h    = (_Float16*)carve(p, (size_t)cVpad * cD * 2); // padded rows
  _Float16* imuW_h   = (_Float16*)carve(p, (size_t)cD * cD * 2);
  _Float16* ilmW_h   = (_Float16*)carve(p, (size_t)cD * cD * 2);
  _Float16* tW1_h    = (_Float16*)carve(p, (size_t)cD * cH * 2);
  _Float16* sW1_h    = (_Float16*)carve(p, (size_t)cS * cD * cF * 2);
  _Float16* sW2_h    = (_Float16*)carve(p, (size_t)cS * cF * cD * 2);
  _Float16* qW_h     = (_Float16*)carve(p, (size_t)cD * cD * 2);
  _Float16* kW_h     = (_Float16*)carve(p, (size_t)cD * cD * 2);
  _Float16* vW_h     = (_Float16*)carve(p, (size_t)cD * cD * 2);
  _Float16* mW1_h    = (_Float16*)carve(p, (size_t)2 * cD * cD * 2);
  _Float16* mW2_h    = (_Float16*)carve(p, (size_t)cD * cD * 2);
  _Float16* oW_h     = (_Float16*)carve(p, (size_t)2 * cD * cD * 2);
  _Float16* wmW_h    = (_Float16*)carve(p, (size_t)2 * cD * cD * 2);
  _Float16* wgW_h    = (_Float16*)carve(p, (size_t)2 * cD * cD * 2);
  _Float16* kact_h   = (_Float16*)carve(p, (size_t)cBT * cD * 2);

  float* h        = (float*)carve(p, (size_t)cBT * cD * 4);   // reused as `final`
  float* mu       = (float*)carve(p, (size_t)cBT * cD * 4);
  float* lam      = (float*)carve(p, (size_t)cBT * cD * 4);
  float* weighted = (float*)carve(p, (size_t)cBT * cD * 4);
  float* sout     = (float*)carve(p, (size_t)cBT * cD * 4);
  float* X1       = (float*)carve(p, (size_t)cBT * cD * 4);
  float* X2       = (float*)carve(p, (size_t)cBT * cD * 4);
  float* local    = (float*)carve(p, (size_t)cBT * cD * 4);
  float* gmsg     = (float*)carve(p, (size_t)cBT * cD * 4);
  float* messages = (float*)carve(p, (size_t)cBT * cD * 4);
  float* mraw     = (float*)carve(p, (size_t)cBT * cD * 4);
  float* kraw     = (float*)carve(p, (size_t)cBT * cD * 4);
  float* qbuf     = (float*)carve(p, (size_t)cBT * cD * 4);
  float* kbuf     = (float*)carve(p, (size_t)cBT * cD * 4);
  float* vbuf     = (float*)carve(p, (size_t)cBT * cD * 4);
  float* tmpF     = (float*)carve(p, (size_t)cBT * cF * 4);   // stage/local hidden
  float* scores   = (float*)carve(p, (size_t)cB * cT * cT * 4);
  float* transhid = (float*)carve(p, (size_t)cBT * cH * 4);
  float* raw49    = (float*)carve(p, (size_t)cBT * 49 * 4);
  float* pi       = (float*)carve(p, (size_t)cBT * cS * 4);
  float* piev     = (float*)carve(p, (size_t)cBT * cS * 4);
  float* evid     = (float*)carve(p, (size_t)cBT * cS * 4);
  float* attn     = (float*)carve(p, (size_t)cBT * cKRET * 4);
  int*   tidx     = (int*)  carve(p, (size_t)cBT * cKRET * 4);

  const float inv_sqrt_d = 1.0f / sqrtf((float)cD);

  // ---- weight f32 -> f16 (once per launch; f16 set ~126MB, L2-resident) ----
  struct CV { _Float16* d; const float* s; long long n; };
  CV cvs[] = {
    {emb_h,  emb,      (long long)cV * cD},
    {imuW_h, init_muW, (long long)cD * cD},
    {ilmW_h, init_lmW, (long long)cD * cD},
    {tW1_h,  transW1,  (long long)cD * cH},
    {sW1_h,  stageW1,  (long long)cS * cD * cF},
    {sW2_h,  stageW2,  (long long)cS * cF * cD},
    {qW_h,   qW,       (long long)cD * cD},
    {kW_h,   kW,       (long long)cD * cD},
    {vW_h,   vW,       (long long)cD * cD},
    {mW1_h,  msgW1,    (long long)2 * cD * cD},
    {mW2_h,  msgW2,    (long long)cD * cD},
    {oW_h,   outW,     (long long)2 * cD * cD},
    {wmW_h,  wmsgW,    (long long)2 * cD * cD},
    {wgW_h,  wgainW,   (long long)2 * cD * cD},
  };
  for (auto& c : cvs) {
    long long n8 = c.n / 8;   // all weight sizes are multiples of 8
    k_convert8<<<EWG(n8), 0, stream>>>(c.d, c.s, n8);
  }
  // zero the padded emb rows [cV, cVpad)
  {
    long long npad = (long long)(cVpad - cV) * cD;
    k_zero_h<<<EWG(npad), 0, stream>>>(emb_h + (long long)cV * cD, npad);
  }

  // ---- init ----
  k_embed<<<EWG((long long)cBT * cD), 0, stream>>>(x, emb, pos_emb, h);
  gemm(stream, h, imuW_h, mu,  cBT, cD, cD, cD, cD, cD, init_mub, 1.0f, 0, 0, 0);
  gemm(stream, h, ilmW_h, lam, cBT, cD, cD, cD, cD, cD, init_lmb, 1.0f, 0, 2, 0);
  k_pi_init<<<EWG(cBT * cS), 0, stream>>>(pi);

  // ---- refinement steps ----
  for (int step = 0; step < cSTEPS; ++step) {
    // content-dependent transition kernel
    gemm(stream, mu, tW1_h, transhid, cBT, cH, cD, cD, cH, cH, transb1, 1.0f, 0, 1, 0);
    k_mm49<<<EWG(cBT * 49), 0, stream>>>(transhid, transW2, transb2, raw49);
    k_transition<<<EWG(cBT), 0, stream>>>(raw49, pi, piev);

    // stage bank (7 experts), weighted by old pi
    for (int s = 0; s < cS; ++s) {
      gemm(stream, mu, sW1_h + (long long)s * cD * cF, tmpF,
           cBT, cF, cD, cD, cF, cF, stageb1 + s * cF, 1.0f, 0, 1, 0);
      gemm(stream, tmpF, sW2_h + (long long)s * cF * cD, sout,
           cBT, cD, cF, cF, cD, cD, stageb2 + s * cD, 1.0f, 0, 0, 0);
      k_stage_acc<<<EWG((long long)cBT * cD), 0, stream>>>(weighted, sout, pi, s, s == 0);
    }
    k_evidence<<<EWG(cBT * cS), 0, stream>>>(weighted, evW, evb, evid);
    k_pi_update<<<EWG(cBT), 0, stream>>>(piev, evid, pi);

    // q/k/v + causal top-8 retrieval
    gemm(stream, mu, qW_h, qbuf, cBT, cD, cD, cD, cD, cD, qb, 1.0f, 0, 0, 0);
    gemm(stream, mu, kW_h, kbuf, cBT, cD, cD, cD, cD, cD, kb, 1.0f, 0, 0, 0);
    gemm(stream, mu, vW_h, vbuf, cBT, cD, cD, cD, cD, cD, vb, 1.0f, 0, 0, 0);
    {
      long long n8 = (long long)cBT * cD / 8;
      k_convert8<<<EWG(n8), 0, stream>>>(kact_h, kbuf, n8);
    }
    gemm(stream, qbuf, kact_h, scores, cT, cT, cD, cD, cD, cT,
         nullptr, inv_sqrt_d, 0, 0, /*btrans=*/1,
         /*Z=*/cB, (long long)cT * cD, (long long)cT * cD, (long long)cT * cT);
    k_topk<<<EWG(cBT), 0, stream>>>(scores, attn, tidx);
    k_gmsg<<<EWG((long long)cBT * cD), 0, stream>>>(attn, tidx, vbuf, gmsg);

    // local windowed messages: cat@W1 = mu@W1_top + shift(mu@W1_bot)
    gemm(stream, mu, mW1_h,                      X1, cBT, cD, cD, cD, cD, cD, msgb1, 1.0f, 0, 0, 0);
    gemm(stream, mu, mW1_h + (long long)cD * cD, X2, cBT, cD, cD, cD, cD, cD, nullptr, 1.0f, 0, 0, 0);
    for (int w = 1; w <= 4; ++w) {
      k_shift_silu<<<EWG((long long)cBT * cD), 0, stream>>>(X1, X2, tmpF, w);
      gemm(stream, tmpF, mW2_h, local, cBT, cD, cD, cD, cD, cD, nullptr, 1.0f, (w > 1), 0, 0);
    }
    k_local_fix<<<EWG((long long)cBT * cD), 0, stream>>>(local, msgb2);

    // messages = ([local, gmsg] @ out_W + out_b) * pi[:,3]
    gemm(stream, local, oW_h,                      messages, cBT, cD, cD, cD, cD, cD, outb, 1.0f, 0, 0, 0);
    gemm(stream, gmsg,  oW_h + (long long)cD * cD, messages, cBT, cD, cD, cD, cD, cD, nullptr, 1.0f, 1, 0, 0);
    k_msg_scale<<<EWG((long long)cBT * cD), 0, stream>>>(messages, pi);

    // Bayesian precision-weighted write
    gemm(stream, mu,       wmW_h,                      mraw, cBT, cD, cD, cD, cD, cD, wmsgb, 1.0f, 0, 0, 0);
    gemm(stream, messages, wmW_h + (long long)cD * cD, mraw, cBT, cD, cD, cD, cD, cD, nullptr, 1.0f, 1, 0, 0);
    gemm(stream, mu,       wgW_h,                      kraw, cBT, cD, cD, cD, cD, cD, wgainb, 1.0f, 0, 0, 0);
    gemm(stream, messages, wgW_h + (long long)cD * cD, kraw, cBT, cD, cD, cD, cD, cD, nullptr, 1.0f, 1, 0, 0);
    k_bayes<<<EWG((long long)cBT * cD), 0, stream>>>(mu, lam, mraw, kraw, pi, weighted);
  }

  // ---- final LayerNorm + tied-embedding readout ----
  k_layernorm<<<EWG(cBT), 0, stream>>>(mu, lng, lnb, h);
  gemm(stream, h, emb_h, logits, cBT, cV, cD, cD, cD, cV,
       nullptr, inv_sqrt_d, 0, 0, /*btrans=*/1);
}
